// StreamingLightDFN_45552423141594
// MI455X (gfx1250) — matrix operand order repaired and checked by reading, not compiled, not run on hardware
//
#include <hip/hip_runtime.h>
#include <hip/hip_bf16.h>

typedef __attribute__((ext_vector_type(16))) _Float16 v16h;
typedef __attribute__((ext_vector_type(8)))  float    v8f;

#define BATCH 1024
#define CCH   64
#define FBINS 481
#define NERB  32
#define NDF   96
#define EMBD  256

__device__ __forceinline__ float sigmf(float x) { return 1.0f / (1.0f + expf(-x)); }

// ---------------------------------------------------------------------------
// WMMA fragment packing (wave32, v_wmma_f32_16x16x32_f16 layouts per CDNA5 ISA)
// A (16x32 f16): lane L -> M = L&15, group g=L>>4; element i (0..15) of v16h:
//   K = (i<8 ? g*8 + i : 16 + g*8 + (i-8)).  Stored 16 halves (32B) per lane.
// B (32x16 f16): lane L -> N = L&15, group g=L>>4; element i: K same as for A.
// ---------------------------------------------------------------------------
__device__ __forceinline__ size_t afrag_off(int m, int k, int K) {
    int kt = k >> 5, kr = k & 31;
    int half = kr >> 4, rem = kr & 15;
    int g = rem >> 3, i = half * 8 + (rem & 7);
    int lane = g * 16 + (m & 15);
    return (((size_t)(m >> 4) * (K >> 5) + kt) * 32 + lane) * 16 + i;
}

__global__ void k_pack_a(const float* __restrict__ X, _Float16* __restrict__ P, int K) {
    int idx = blockIdx.x * blockDim.x + threadIdx.x;
    if (idx >= BATCH * K) return;
    int b = idx / K, k = idx % K;
    P[afrag_off(b, k, K)] = (_Float16)X[idx];
}

__global__ void k_pack_b(const float* __restrict__ W, _Float16* __restrict__ P, int K, int N) {
    int idx = blockIdx.x * blockDim.x + threadIdx.x;
    if (idx >= K * N) return;
    int k = idx / N, n = idx % N;
    int g = (k >> 3) & 1, i = ((k >> 4) & 1) * 8 + (k & 7);
    int lane = g * 16 + (n & 15);
    size_t off = (((size_t)(k >> 5) * (N >> 4) + (n >> 4)) * 32 + lane) * 16 + i;
    P[off] = (_Float16)W[idx];
}

// pack A for FC input: k = f*64 + ci over c1[B,64,48] (transpose flatten)
__global__ void k_pack_c1(const float* __restrict__ c1, _Float16* __restrict__ P) {
    int idx = blockIdx.x * blockDim.x + threadIdx.x;
    if (idx >= BATCH * 3072) return;
    int b = idx / 3072, k = idx % 3072;
    int f = k >> 6, ci = k & 63;
    P[afrag_off(b, k, 3072)] = (_Float16)c1[((size_t)b * CCH + ci) * 48 + f];
}

// emb = e3_flat + cemb  -> packed A fragments
__global__ void k_emb_pack(const float* __restrict__ cemb, const float* __restrict__ e3,
                           _Float16* __restrict__ P) {
    int idx = blockIdx.x * blockDim.x + threadIdx.x;
    if (idx >= BATCH * EMBD) return;
    int b = idx / EMBD, k = idx % EMBD;
    int f = k >> 6, ci = k & 63;
    float v = cemb[idx] + e3[((size_t)b * CCH + ci) * 4 + f];
    P[afrag_off(b, k, EMBD)] = (_Float16)v;
}

// c = ch + skip -> packed A fragments
__global__ void k_add_pack(const float* __restrict__ ch, const float* __restrict__ sk,
                           _Float16* __restrict__ P) {
    int idx = blockIdx.x * blockDim.x + threadIdx.x;
    if (idx >= BATCH * EMBD) return;
    int b = idx / EMBD, j = idx % EMBD;
    P[afrag_off(b, j, EMBD)] = (_Float16)(ch[idx] + sk[idx]);
}

// ---------------------------------------------------------------------------
// GEMM: C[MxN] = act(A[MxK] @ B[KxN] + bias), A/B pre-packed f16 fragments.
// One wave -> 16x16 tile; 4 waves/block cover a 16x64 strip.
// ---------------------------------------------------------------------------
__global__ void k_wmma_gemm(const _Float16* __restrict__ Ap, const _Float16* __restrict__ Bp,
                            const float* __restrict__ bias, float* __restrict__ C,
                            int M, int N, int K, int act) {
    const int lane = threadIdx.x & 31;
    const int nt = blockIdx.x * 4 + (threadIdx.x >> 5);
    const int mt = blockIdx.y;
    const int NT = N >> 4, KT = K >> 5;
    if (nt >= NT || mt * 16 >= M) return;
    v8f acc = {};
    const _Float16* ap = Ap + ((size_t)mt * KT * 32 + lane) * 16;
    const _Float16* bp = Bp + ((size_t)nt * 32 + lane) * 16;
    for (int kt = 0; kt < KT; ++kt) {
        v16h a = *(const v16h*)ap;  ap += 512;
        v16h b = *(const v16h*)bp;  bp += (size_t)NT * 512;
        acc = __builtin_amdgcn_wmma_f32_16x16x32_f16(false, a, false, b,
                                                     (short)0, acc, false, false);
    }
    const int g = lane >> 4, col = lane & 15;
    const int n = nt * 16 + col;
    const float bv = bias ? bias[n] : 0.0f;
#pragma unroll
    for (int r = 0; r < 8; ++r) {
        int m = mt * 16 + 8 * g + r;
        float v = acc[r] + bv;
        if (act) v = fmaxf(v, 0.0f);
        C[(size_t)m * N + n] = v;
    }
}

// ---------------------------------------------------------------------------
// Encoder front convs
// ---------------------------------------------------------------------------
__global__ void k_conv_erb0(const float* __restrict__ buf, const float* __restrict__ feat,
                            const float* __restrict__ W, const float* __restrict__ bias,
                            float* __restrict__ e0) {
    int idx = blockIdx.x * blockDim.x + threadIdx.x;
    if (idx >= BATCH * CCH * NERB) return;
    int f = idx % NERB, c = (idx / NERB) % CCH, b = idx / (NERB * CCH);
    const float* x0 = buf + (size_t)b * 2 * NERB;
    const float* x2 = feat + (size_t)b * NERB;
    const float* w = W + c * 9;
    float acc = bias[c];
    for (int kf = 0; kf < 3; ++kf) {
        int ff = f + kf - 1;
        if (ff < 0 || ff >= NERB) continue;
        acc += x0[ff] * w[kf] + x0[NERB + ff] * w[3 + kf] + x2[ff] * w[6 + kf];
    }
    e0[idx] = fmaxf(acc, 0.0f);
}

__global__ void k_conv_df0(const float* __restrict__ buf, const float* __restrict__ fs,
                           const float* __restrict__ W, const float* __restrict__ bias,
                           float* __restrict__ c0) {
    int idx = blockIdx.x * blockDim.x + threadIdx.x;
    if (idx >= BATCH * CCH * NDF) return;
    int f = idx % NDF, c = (idx / NDF) % CCH, b = idx / (NDF * CCH);
    float acc = bias[c];
    for (int ci = 0; ci < 2; ++ci) {
        const float* xb = buf + ((size_t)b * 2 + ci) * 2 * NDF;
        const float* w = W + (c * 2 + ci) * 9;
        for (int kf = 0; kf < 3; ++kf) {
            int ff = f + kf - 1;
            if (ff < 0 || ff >= NDF) continue;
            acc += xb[ff] * w[kf] + xb[NDF + ff] * w[3 + kf]
                 + fs[((size_t)b * NDF + ff) * 2 + ci] * w[6 + kf];
        }
    }
    c0[idx] = fmaxf(acc, 0.0f);
}

// C->C conv, kernel (1,3), stride 2, pad 1, relu. Weights staged in LDS.
__global__ void k_conv_down(const float* __restrict__ X, const float* __restrict__ W,
                            const float* __restrict__ bias, float* __restrict__ Y,
                            int Fin, int Fout) {
    __shared__ float sw[CCH * CCH * 3];
    for (int i = threadIdx.x; i < CCH * CCH * 3; i += blockDim.x) sw[i] = W[i];
    __syncthreads();
    int idx = blockIdx.x * blockDim.x + threadIdx.x;
    if (idx >= BATCH * CCH * Fout) return;
    int f = idx % Fout, co = (idx / Fout) % CCH, b = idx / (Fout * CCH);
    const float* xb = X + (size_t)b * CCH * Fin;
    const float* wc = sw + co * CCH * 3;
    float acc = bias[co];
    int f0 = 2 * f - 1;
    for (int ci = 0; ci < CCH; ++ci) {
        const float* xr = xb + ci * Fin;
        const float* wr = wc + ci * 3;
        float s = xr[f0 + 1] * wr[1];
        if (f0 >= 0) s += xr[f0] * wr[0];
        if (f0 + 2 < Fin) s += xr[f0 + 2] * wr[2];
        acc += s;
    }
    Y[idx] = fmaxf(acc, 0.0f);
}

// 1x1 conv + relu, then optional decoder-add and optional emb_erb add
__global__ void k_path(const float* __restrict__ X, const float* __restrict__ W,
                       const float* __restrict__ bias, const float* __restrict__ addD,
                       const float* __restrict__ embErb, float* __restrict__ Y, int F) {
    __shared__ float sw[CCH * CCH];
    for (int i = threadIdx.x; i < CCH * CCH; i += blockDim.x) sw[i] = W[i];
    __syncthreads();
    int idx = blockIdx.x * blockDim.x + threadIdx.x;
    if (idx >= BATCH * CCH * F) return;
    int f = idx % F, co = (idx / F) % CCH, b = idx / (F * CCH);
    const float* xb = X + (size_t)b * CCH * F;
    const float* wc = sw + co * CCH;
    float acc = bias[co];
    for (int ci = 0; ci < CCH; ++ci) acc += xb[ci * F + f] * wc[ci];
    float v = fmaxf(acc, 0.0f);
    if (addD) v += addD[idx];
    if (embErb) v += embErb[(size_t)b * EMBD + f * CCH + co];
    Y[idx] = v;
}

// ConvTranspose (1,3) stride (1,2) pad 1 outpad 1: Fo = 2*Fin. relu.
__global__ void k_convt(const float* __restrict__ X, const float* __restrict__ W,
                        const float* __restrict__ bias, float* __restrict__ Y, int Fin) {
    __shared__ float sw[CCH * CCH * 3];
    for (int i = threadIdx.x; i < CCH * CCH * 3; i += blockDim.x) sw[i] = W[i];
    __syncthreads();
    int Fo = 2 * Fin;
    int idx = blockIdx.x * blockDim.x + threadIdx.x;
    if (idx >= BATCH * CCH * Fo) return;
    int of = idx % Fo, co = (idx / Fo) % CCH, b = idx / (Fo * CCH);
    const float* xb = X + (size_t)b * CCH * Fin;
    const float* wc = sw + co * CCH * 3;
    float acc = bias[co];
    if ((of & 1) == 0) {
        int fi = of >> 1;
        for (int ci = 0; ci < CCH; ++ci) acc += xb[ci * Fin + fi] * wc[ci * 3 + 1];
    } else {
        int fl = (of - 1) >> 1, fr = (of + 1) >> 1;
        bool hasR = fr < Fin;
        for (int ci = 0; ci < CCH; ++ci) {
            float s = xb[ci * Fin + fl] * wc[ci * 3 + 2];
            if (hasR) s += xb[ci * Fin + fr] * wc[ci * 3 + 0];
            acc += s;
        }
    }
    Y[idx] = fmaxf(acc, 0.0f);
}

__global__ void k_maskconv(const float* __restrict__ s0, const float* __restrict__ W,
                           const float* __restrict__ bout, float* __restrict__ mask) {
    int idx = blockIdx.x * blockDim.x + threadIdx.x;
    if (idx >= BATCH * NERB) return;
    int f = idx % NERB, b = idx / NERB;
    const float* xb = s0 + (size_t)b * CCH * NERB;
    float acc = bout[0];
    for (int ci = 0; ci < CCH; ++ci) {
        const float* w = W + ci * 3;
        for (int kf = 0; kf < 3; ++kf) {
            int ff = f + kf - 1;
            if (ff >= 0 && ff < NERB) acc += xb[ci * NERB + ff] * w[kf];
        }
    }
    mask[idx] = sigmf(acc);
}

__global__ void k_gain(const float* __restrict__ mask, const float* __restrict__ erb_inv,
                       const float* __restrict__ spec, float* __restrict__ outspec) {
    int idx = blockIdx.x * blockDim.x + threadIdx.x;
    if (idx >= BATCH * FBINS) return;
    int f = idx % FBINS, b = idx / FBINS;
    const float* mb = mask + (size_t)b * NERB;
    float g = 0.0f;
    for (int e = 0; e < NERB; ++e) g += mb[e] * erb_inv[e * FBINS + f];
    size_t so = ((size_t)b * FBINS + f) * 2;
    outspec[so]     = spec[so] * g;
    outspec[so + 1] = spec[so + 1] * g;
}

// ---------------------------------------------------------------------------
// GRU pointwise
// ---------------------------------------------------------------------------
__global__ void k_gru(const float* __restrict__ gi, const float* __restrict__ gh,
                      const float* __restrict__ h, float* __restrict__ hnew,
                      _Float16* __restrict__ pack) {
    int idx = blockIdx.x * blockDim.x + threadIdx.x;
    if (idx >= BATCH * EMBD) return;
    int b = idx / EMBD, j = idx % EMBD;
    const float* gib = gi + (size_t)b * 3 * EMBD;
    const float* ghb = gh + (size_t)b * 3 * EMBD;
    float r = sigmf(gib[j] + ghb[j]);
    float z = sigmf(gib[EMBD + j] + ghb[EMBD + j]);
    float n = tanhf(gib[2 * EMBD + j] + r * ghb[2 * EMBD + j]);
    float o = (1.0f - z) * n + z * h[idx];
    hnew[idx] = o;
    if (pack) pack[afrag_off(b, j, EMBD)] = (_Float16)o;
}

// ---------------------------------------------------------------------------
// DF branch: temporal conv (5,1) over buf_dfp||c0, combine, complex filter
// ---------------------------------------------------------------------------
__global__ void k_c0d(const float* __restrict__ bufdfp, const float* __restrict__ c0,
                      const float* __restrict__ W, const float* __restrict__ bias,
                      float* __restrict__ Y) {
    int idx = blockIdx.x * blockDim.x + threadIdx.x;
    if (idx >= BATCH * 10 * NDF) return;
    int f = idx % NDF, o = (idx / NDF) % 10, b = idx / (NDF * 10);
    float acc = bias[o];
    const float* wb = W + o * CCH * 5;
    for (int ci = 0; ci < CCH; ++ci) {
        const float* xr = bufdfp + ((size_t)b * CCH + ci) * 4 * NDF + f;
        const float* w = wb + ci * 5;
        acc += xr[0] * w[0] + xr[NDF] * w[1] + xr[2 * NDF] * w[2] + xr[3 * NDF] * w[3];
        acc += c0[((size_t)b * CCH + ci) * NDF + f] * w[4];
    }
    Y[idx] = acc;
}

__global__ void k_df(const float* __restrict__ dfout, const float* __restrict__ c0d,
                     const float* __restrict__ bufspec, const float* __restrict__ spec,
                     float* __restrict__ outspec) {
    int idx = blockIdx.x * blockDim.x + threadIdx.x;
    if (idx >= BATCH * NDF) return;
    int f = idx % NDF, b = idx / NDF;
    float rr = 0.0f, ri = 0.0f;
    for (int n = 0; n < 5; ++n) {
        float cr = dfout[(size_t)b * 960 + f * 10 + n * 2]
                 + c0d[((size_t)b * 10 + n * 2) * NDF + f];
        float ci = dfout[(size_t)b * 960 + f * 10 + n * 2 + 1]
                 + c0d[((size_t)b * 10 + n * 2 + 1) * NDF + f];
        float sr, si;
        if (n < 4) {
            size_t o = (((size_t)b * 4 + n) * FBINS + f) * 2;
            sr = bufspec[o]; si = bufspec[o + 1];
        } else {
            size_t o = ((size_t)b * FBINS + f) * 2;
            sr = spec[o]; si = spec[o + 1];
        }
        rr += sr * cr - si * ci;
        ri += sr * ci + si * cr;
    }
    size_t so = ((size_t)b * FBINS + f) * 2;
    outspec[so] = rr; outspec[so + 1] = ri;
}

// ---------------------------------------------------------------------------
// State buffer shifts
// ---------------------------------------------------------------------------
__global__ void k_shift_erb0(const float* __restrict__ buf, const float* __restrict__ feat,
                             float* __restrict__ out) {
    int idx = blockIdx.x * blockDim.x + threadIdx.x;
    if (idx >= BATCH * NERB) return;
    int f = idx % NERB, b = idx / NERB;
    out[((size_t)b * 2 + 0) * NERB + f] = buf[((size_t)b * 2 + 1) * NERB + f];
    out[((size_t)b * 2 + 1) * NERB + f] = feat[(size_t)b * NERB + f];
}

__global__ void k_shift_df0(const float* __restrict__ buf, const float* __restrict__ fs,
                            float* __restrict__ out) {
    int idx = blockIdx.x * blockDim.x + threadIdx.x;
    if (idx >= BATCH * 2 * NDF) return;
    int f = idx % NDF, ci = (idx / NDF) & 1, b = idx / (2 * NDF);
    size_t base = ((size_t)b * 2 + ci) * 2 * NDF;
    out[base + f]       = buf[base + NDF + f];
    out[base + NDF + f] = fs[((size_t)b * NDF + f) * 2 + ci];
}

__global__ void k_shift_dfp(const float* __restrict__ buf, const float* __restrict__ c0,
                            float* __restrict__ out) {
    int idx = blockIdx.x * blockDim.x + threadIdx.x;
    if (idx >= BATCH * CCH * NDF) return;
    int f = idx % NDF, c = (idx / NDF) % CCH, b = idx / (NDF * CCH);
    size_t base = ((size_t)b * CCH + c) * 4 * NDF;
    out[base + f]           = buf[base + NDF + f];
    out[base + NDF + f]     = buf[base + 2 * NDF + f];
    out[base + 2 * NDF + f] = buf[base + 3 * NDF + f];
    out[base + 3 * NDF + f] = c0[((size_t)b * CCH + c) * NDF + f];
}

__global__ void k_shift_spec(const float* __restrict__ buf, const float* __restrict__ spec,
                             float* __restrict__ out) {
    int idx = blockIdx.x * blockDim.x + threadIdx.x;
    if (idx >= BATCH * FBINS * 2) return;
    int ri = idx & 1, f = (idx >> 1) % FBINS, b = idx / (FBINS * 2);
    size_t bs = (size_t)b * 4 * FBINS * 2;
    for (int n = 0; n < 3; ++n)
        out[bs + (((size_t)n * FBINS + f) * 2) + ri] =
            buf[bs + (((size_t)(n + 1) * FBINS + f) * 2) + ri];
    out[bs + (((size_t)3 * FBINS + f) * 2) + ri] = spec[((size_t)b * FBINS + f) * 2 + ri];
}

// ---------------------------------------------------------------------------
extern "C" void kernel_launch(void* const* d_in, const int* in_sizes, int n_in,
                              void* d_out, int out_size, void* d_ws, size_t ws_size,
                              hipStream_t stream) {
    const float* spec      = (const float*)d_in[0];
    const float* feat_erb  = (const float*)d_in[1];
    const float* feat_spec = (const float*)d_in[2];
    const float* h_enc     = (const float*)d_in[3];
    const float* h_erb     = (const float*)d_in[4];
    const float* h_df      = (const float*)d_in[5];
    const float* buf_erb0  = (const float*)d_in[6];
    const float* buf_df0   = (const float*)d_in[7];
    const float* buf_dfp   = (const float*)d_in[8];
    const float* buf_spec  = (const float*)d_in[9];
    const float* w_erb0    = (const float*)d_in[10];
    const float* b_erb0    = (const float*)d_in[11];
    const float* w_erb123  = (const float*)d_in[12];
    const float* b_erb123  = (const float*)d_in[13];
    const float* w_df0     = (const float*)d_in[14];
    const float* b_df0     = (const float*)d_in[15];
    const float* w_df1     = (const float*)d_in[16];
    const float* b_df1     = (const float*)d_in[17];
    const float* w_fc      = (const float*)d_in[18];
    const float* b_fc      = (const float*)d_in[19];
    const float* gwi_enc   = (const float*)d_in[20];
    const float* gwh_enc   = (const float*)d_in[21];
    const float* gbi_enc   = (const float*)d_in[22];
    const float* gbh_enc   = (const float*)d_in[23];
    const float* gwi_erb   = (const float*)d_in[24];
    const float* gwh_erb   = (const float*)d_in[25];
    const float* gbi_erb   = (const float*)d_in[26];
    const float* gbh_erb   = (const float*)d_in[27];
    const float* gwi_df    = (const float*)d_in[28];
    const float* gwh_df    = (const float*)d_in[29];
    const float* gbi_df    = (const float*)d_in[30];
    const float* gbh_df    = (const float*)d_in[31];
    const float* w_skip    = (const float*)d_in[32];
    const float* b_skip    = (const float*)d_in[33];
    const float* w_dfout   = (const float*)d_in[34];
    const float* b_dfout   = (const float*)d_in[35];
    const float* w_path    = (const float*)d_in[36];
    const float* b_path    = (const float*)d_in[37];
    const float* w_convt   = (const float*)d_in[38];
    const float* b_convt   = (const float*)d_in[39];
    const float* w_out     = (const float*)d_in[40];
    const float* b_out     = (const float*)d_in[41];
    const float* w_dfp     = (const float*)d_in[42];
    const float* b_dfp     = (const float*)d_in[43];
    const float* erb_inv   = (const float*)d_in[44];

    float* out = (float*)d_out;
    const size_t o_henc  = (size_t)BATCH * FBINS * 2;
    const size_t o_herb  = o_henc + (size_t)BATCH * EMBD;
    const size_t o_hdf   = o_herb + (size_t)BATCH * EMBD;
    const size_t o_berb  = o_hdf  + (size_t)BATCH * EMBD;
    const size_t o_bdf0  = o_berb + (size_t)BATCH * 2 * NERB;
    const size_t o_bdfp  = o_bdf0 + (size_t)BATCH * 2 * 2 * NDF;

    // bump allocator over workspace
    char* wsp = (char*)d_ws;
    size_t cur = 0;
    auto alloc = [&](size_t bytes) -> void* {
        void* p = wsp + cur;
        cur = (cur + bytes + 255) & ~(size_t)255;
        return p;
    };

    float* e0    = (float*)alloc((size_t)BATCH * CCH * NERB * 4);
    float* c0    = (float*)alloc((size_t)BATCH * CCH * NDF * 4);
    float* e1    = (float*)alloc((size_t)BATCH * CCH * 16 * 4);
    float* e2    = (float*)alloc((size_t)BATCH * CCH * 8 * 4);
    float* e3    = (float*)alloc((size_t)BATCH * CCH * 4 * 4);
    float* c1    = (float*)alloc((size_t)BATCH * CCH * 48 * 4);
    _Float16* afc  = (_Float16*)alloc((size_t)BATCH * 3072 * 2);
    float* cemb  = (float*)alloc((size_t)BATCH * EMBD * 4);
    _Float16* embp = (_Float16*)alloc((size_t)BATCH * EMBD * 2);
    _Float16* hp   = (_Float16*)alloc((size_t)BATCH * EMBD * 2);
    _Float16* emb2p= (_Float16*)alloc((size_t)BATCH * EMBD * 2);
    _Float16* cpk  = (_Float16*)alloc((size_t)BATCH * EMBD * 2);
    float* gi    = (float*)alloc((size_t)BATCH * 768 * 4);
    float* gh    = (float*)alloc((size_t)BATCH * 768 * 4);
    float* skipb = (float*)alloc((size_t)BATCH * EMBD * 4);
    float* dfo   = (float*)alloc((size_t)BATCH * 960 * 4);
    float* s3    = (float*)alloc((size_t)BATCH * CCH * 4 * 4);
    float* e3d   = (float*)alloc((size_t)BATCH * CCH * 8 * 4);
    float* s2    = (float*)alloc((size_t)BATCH * CCH * 8 * 4);
    float* e2d   = (float*)alloc((size_t)BATCH * CCH * 16 * 4);
    float* s1    = (float*)alloc((size_t)BATCH * CCH * 16 * 4);
    float* e1d   = (float*)alloc((size_t)BATCH * CCH * NERB * 4);
    float* s0    = (float*)alloc((size_t)BATCH * CCH * NERB * 4);
    float* maskb = (float*)alloc((size_t)BATCH * NERB * 4);
    float* c0d   = (float*)alloc((size_t)BATCH * 10 * NDF * 4);
    _Float16* wfc_p = (_Float16*)alloc((size_t)3072 * 256 * 2);
    _Float16* wie_p = (_Float16*)alloc((size_t)256 * 768 * 2);
    _Float16* whe_p = (_Float16*)alloc((size_t)256 * 768 * 2);
    _Float16* wir_p = (_Float16*)alloc((size_t)256 * 768 * 2);
    _Float16* whr_p = (_Float16*)alloc((size_t)256 * 768 * 2);
    _Float16* wid_p = (_Float16*)alloc((size_t)256 * 768 * 2);
    _Float16* whd_p = (_Float16*)alloc((size_t)256 * 768 * 2);
    _Float16* wsk_p = (_Float16*)alloc((size_t)256 * 256 * 2);
    _Float16* wdo_p = (_Float16*)alloc((size_t)256 * 960 * 2);

    const dim3 TPB(256);
#define NBLK(n) dim3((unsigned)(((n) + 255) / 256))
    auto gemm = [&](const _Float16* A, const _Float16* B, const float* bias,
                    float* C, int M, int N, int K, int act) {
        dim3 grid((unsigned)((N / 16 + 3) / 4), (unsigned)(M / 16));
        k_wmma_gemm<<<grid, dim3(128), 0, stream>>>(A, B, bias, C, M, N, K, act);
    };

    // --- pack weights to f16 WMMA fragments (per call; deterministic) ---
    k_pack_b<<<NBLK(3072 * 256), TPB, 0, stream>>>(w_fc,    wfc_p, 3072, 256);
    k_pack_b<<<NBLK(256 * 768),  TPB, 0, stream>>>(gwi_enc, wie_p, 256, 768);
    k_pack_b<<<NBLK(256 * 768),  TPB, 0, stream>>>(gwh_enc, whe_p, 256, 768);
    k_pack_b<<<NBLK(256 * 768),  TPB, 0, stream>>>(gwi_erb, wir_p, 256, 768);
    k_pack_b<<<NBLK(256 * 768),  TPB, 0, stream>>>(gwh_erb, whr_p, 256, 768);
    k_pack_b<<<NBLK(256 * 768),  TPB, 0, stream>>>(gwi_df,  wid_p, 256, 768);
    k_pack_b<<<NBLK(256 * 768),  TPB, 0, stream>>>(gwh_df,  whd_p, 256, 768);
    k_pack_b<<<NBLK(256 * 256),  TPB, 0, stream>>>(w_skip,  wsk_p, 256, 256);
    k_pack_b<<<NBLK(256 * 960),  TPB, 0, stream>>>(w_dfout, wdo_p, 256, 960);

    // --- encoder ---
    k_conv_erb0<<<NBLK(BATCH * CCH * NERB), TPB, 0, stream>>>(buf_erb0, feat_erb, w_erb0, b_erb0, e0);
    k_conv_df0<<<NBLK(BATCH * CCH * NDF),  TPB, 0, stream>>>(buf_df0, feat_spec, w_df0, b_df0, c0);
    k_conv_down<<<NBLK(BATCH * CCH * 16), TPB, 0, stream>>>(e0, w_erb123,               b_erb123,       e1, 32, 16);
    k_conv_down<<<NBLK(BATCH * CCH * 8),  TPB, 0, stream>>>(e1, w_erb123 + 12288,       b_erb123 + 64,  e2, 16, 8);
    k_conv_down<<<NBLK(BATCH * CCH * 4),  TPB, 0, stream>>>(e2, w_erb123 + 24576,       b_erb123 + 128, e3, 8, 4);
    k_conv_down<<<NBLK(BATCH * CCH * 48), TPB, 0, stream>>>(c0, w_df1,                  b_df1,          c1, 96, 48);

    k_pack_c1<<<NBLK(BATCH * 3072), TPB, 0, stream>>>(c1, afc);
    gemm(afc, wfc_p, b_fc, cemb, BATCH, 256, 3072, 1);            // cemb = relu(fc)
    k_emb_pack<<<NBLK(BATCH * EMBD), TPB, 0, stream>>>(cemb, e3, embp);

    // --- GRU enc ---
    k_pack_a<<<NBLK(BATCH * EMBD), TPB, 0, stream>>>(h_enc, hp, EMBD);
    gemm(embp, wie_p, gbi_enc, gi, BATCH, 768, 256, 0);
    gemm(hp,   whe_p, gbh_enc, gh, BATCH, 768, 256, 0);
    k_gru<<<NBLK(BATCH * EMBD), TPB, 0, stream>>>(gi, gh, h_enc, out + o_henc, emb2p);

    // --- GRU erb ---
    k_pack_a<<<NBLK(BATCH * EMBD), TPB, 0, stream>>>(h_erb, hp, EMBD);
    gemm(emb2p, wir_p, gbi_erb, gi, BATCH, 768, 256, 0);
    gemm(hp,    whr_p, gbh_erb, gh, BATCH, 768, 256, 0);
    k_gru<<<NBLK(BATCH * EMBD), TPB, 0, stream>>>(gi, gh, h_erb, out + o_herb, (_Float16*)nullptr);

    // --- GRU df + skip + dfout ---
    k_pack_a<<<NBLK(BATCH * EMBD), TPB, 0, stream>>>(h_df, hp, EMBD);
    gemm(emb2p, wid_p, gbi_df, gi, BATCH, 768, 256, 0);
    gemm(hp,    whd_p, gbh_df, gh, BATCH, 768, 256, 0);
    k_gru<<<NBLK(BATCH * EMBD), TPB, 0, stream>>>(gi, gh, h_df, out + o_hdf, (_Float16*)nullptr);
    gemm(emb2p, wsk_p, b_skip, skipb, BATCH, 256, 256, 0);
    k_add_pack<<<NBLK(BATCH * EMBD), TPB, 0, stream>>>(out + o_hdf, skipb, cpk);
    gemm(cpk, wdo_p, b_dfout, dfo, BATCH, 960, 256, 0);

    // --- ERB decoder ---
    k_path<<<NBLK(BATCH * CCH * 4),  TPB, 0, stream>>>(e3, w_path + 3 * 4096, b_path + 192, nullptr, out + o_herb, s3, 4);
    k_convt<<<NBLK(BATCH * CCH * 8), TPB, 0, stream>>>(s3, w_convt + 24576, b_convt + 128, e3d, 4);
    k_path<<<NBLK(BATCH * CCH * 8),  TPB, 0, stream>>>(e2, w_path + 2 * 4096, b_path + 128, e3d, nullptr, s2, 8);
    k_convt<<<NBLK(BATCH * CCH * 16), TPB, 0, stream>>>(s2, w_convt + 12288, b_convt + 64, e2d, 8);
    k_path<<<NBLK(BATCH * CCH * 16), TPB, 0, stream>>>(e1, w_path + 4096, b_path + 64, e2d, nullptr, s1, 16);
    k_convt<<<NBLK(BATCH * CCH * 32), TPB, 0, stream>>>(s1, w_convt, b_convt, e1d, 16);
    k_path<<<NBLK(BATCH * CCH * 32), TPB, 0, stream>>>(e0, w_path, b_path, e1d, nullptr, s0, 32);
    k_maskconv<<<NBLK(BATCH * NERB), TPB, 0, stream>>>(s0, w_out, b_out, maskb);
    k_gain<<<NBLK(BATCH * FBINS), TPB, 0, stream>>>(maskb, erb_inv, spec, out);

    // --- DF branch (overwrites first NB_DF bins of spec_e) ---
    k_c0d<<<NBLK(BATCH * 10 * NDF), TPB, 0, stream>>>(buf_dfp, c0, w_dfp, b_dfp, c0d);
    k_df<<<NBLK(BATCH * NDF), TPB, 0, stream>>>(dfo, c0d, buf_spec, spec, out);

    // --- state buffer shifts ---
    k_shift_erb0<<<NBLK(BATCH * NERB), TPB, 0, stream>>>(buf_erb0, feat_erb, out + o_berb);
    k_shift_df0<<<NBLK(BATCH * 2 * NDF), TPB, 0, stream>>>(buf_df0, feat_spec, out + o_bdf0);
    k_shift_dfp<<<NBLK(BATCH * CCH * NDF), TPB, 0, stream>>>(buf_dfp, c0, out + o_bdfp);
    k_shift_spec<<<NBLK(BATCH * FBINS * 2), TPB, 0, stream>>>(buf_spec, spec, out + o_bdfp + (size_t)BATCH * CCH * 4 * NDF);

    (void)in_sizes; (void)n_in; (void)out_size; (void)ws_size;
#undef NBLK
}
